// Attention_48241072668890
// MI455X (gfx1250) — compile-verified, matrix operand order checked by reading, not run on hardware
//
#include <hip/hip_runtime.h>

// ---------------------------------------------------------------------------
// Attention block for MI455X (gfx1250, wave32, WMMA).
// Live dataflow:  q = x@wq^T ; flash-attn(q, cache_k, cache_v, mask) ; @wo^T
// Operands pre-converted fp32->bf16 once; hot loops use async global->LDS
// copies (ASYNCcnt) double-buffered against v_wmma_f32_16x16x32_bf16.
// ---------------------------------------------------------------------------

#define DIM   4096
#define NH    32
#define HD    128
#define SEQ   1024
#define CTX   2048
#define NBAT  2

typedef __attribute__((ext_vector_type(16))) __bf16       bf16x16;
typedef __attribute__((ext_vector_type(8)))  float        floatx8;
typedef __attribute__((ext_vector_type(4)))  float        f32x4;
typedef __attribute__((ext_vector_type(4)))  unsigned int u32x4;

__device__ __forceinline__ unsigned short f2bf(float f) {
  unsigned u = __float_as_uint(f);
  u += 0x7FFFu + ((u >> 16) & 1u);          // round-to-nearest-even
  return (unsigned short)(u >> 16);
}

// 16B async copy: global -> LDS, tracked by ASYNCcnt (per-lane addresses).
__device__ __forceinline__ void async_cp16(unsigned lds_byte_addr, const void* gptr) {
  asm volatile("global_load_async_to_lds_b128 %0, %1, off"
               :: "v"(lds_byte_addr), "v"((unsigned long long)(size_t)gptr)
               : "memory");
}
#define WAIT_ASYNC(n) asm volatile("s_wait_asynccnt " #n ::: "memory")

// A-style 16x32 bf16 fragment from an LDS tile stored row-major.
// Per ISA 7.12.2: lane&15 = row M ; lane>=16 selects K-offset 8 ;
// regs 0..3 = K in [koff,koff+8) ; regs 4..7 = K in [koff+16,koff+24).
__device__ __forceinline__ bf16x16 load_frag(const unsigned short* tile, int stride) {
  const int lane = threadIdx.x & 31;
  const int row  = lane & 15;
  const int koff = (lane >> 4) << 3;
  union { bf16x16 v; u32x4 q[2]; } f;
  f.q[0] = *(const u32x4*)(tile + row * stride + koff);
  f.q[1] = *(const u32x4*)(tile + row * stride + koff + 16);
  return f.v;
}

#define WMMA_BF16(Afr, Bfr, Cfr) \
  __builtin_amdgcn_wmma_f32_16x16x32_bf16(false, (Afr), false, (Bfr), (short)0, (Cfr), false, false)

// ---------------------------------------------------------------------------
// fp32 -> bf16 elementwise convert (n multiple of 8)
// ---------------------------------------------------------------------------
__global__ __launch_bounds__(256)
void cvt_f32_bf16(const float* __restrict__ src, unsigned short* __restrict__ dst, int n) {
  const int i = (blockIdx.x * 256 + threadIdx.x) * 8;
  if (i >= n) return;
  f32x4 a = *(const f32x4*)(src + i);
  f32x4 b = *(const f32x4*)(src + i + 4);
  union { u32x4 q; unsigned short e[8]; } o;
  for (int j = 0; j < 4; ++j) o.e[j]     = f2bf(a[j]);
  for (int j = 0; j < 4; ++j) o.e[4 + j] = f2bf(b[j]);
  *(u32x4*)(dst + i) = o.q;
}

// ---------------------------------------------------------------------------
// cache_v [B,C,H,HD] fp32  ->  Vt [B,H,HD,CTX] bf16 (per-head transpose)
// ---------------------------------------------------------------------------
__global__ __launch_bounds__(256)
void cvt_transpose_v(const float* __restrict__ V, unsigned short* __restrict__ Vt) {
  const int b  = blockIdx.x >> 5;
  const int h  = blockIdx.x & 31;
  const int c0 = blockIdx.y * 32;
  const int e0 = blockIdx.z * 32;
  __shared__ unsigned short t[32][33];
  const int tid = threadIdx.x;
  {
    const int c = tid >> 3, e4 = (tid & 7) << 2;
    f32x4 v = *(const f32x4*)(V + ((size_t)(b * CTX + c0 + c) * NH + h) * HD + e0 + e4);
    for (int j = 0; j < 4; ++j) t[c][e4 + j] = f2bf(v[j]);
  }
  __syncthreads();
  {
    const int e = tid >> 3, c4 = (tid & 7) << 2;
    union { unsigned long long q; unsigned short s[4]; } o;
    for (int j = 0; j < 4; ++j) o.s[j] = t[c4 + j][e];
    *(unsigned long long*)(Vt + ((size_t)(b * NH + h) * HD + e0 + e) * CTX + c0 + c4) = o.q;
  }
}

// ---------------------------------------------------------------------------
// GEMM: C[M,N] = A[M,K] * W[N,K]^T, bf16 in, fp32 accum, fp32 or bf16 out.
// Block tile 128x128, K-step 32; async-to-LDS double buffering.
// ---------------------------------------------------------------------------
#define TM  128
#define TN  128
#define TK  32
#define LDT 40      // 32 + 8 pad ushorts -> 80B rows (16B aligned)

template <bool BF16_OUT>
__global__ __launch_bounds__(256)
void gemm_bf16(const unsigned short* __restrict__ A, const unsigned short* __restrict__ W,
               float* __restrict__ Cf, unsigned short* __restrict__ Cb,
               int M, int N, int K) {
  __shared__ unsigned short As[2][TM * LDT];
  __shared__ unsigned short Ws[2][TN * LDT];

  const int tid  = threadIdx.x;
  const int lane = tid & 31;
  const int wave = tid >> 5;
  const int wm   = wave >> 1;                 // 0..3 : 32-row group
  const int wn   = wave & 1;                  // 0..1 : 64-col group
  const int bm   = blockIdx.x * TM;
  const int bn   = blockIdx.y * TN;

  const floatx8 fz = {0.f,0.f,0.f,0.f,0.f,0.f,0.f,0.f};
  floatx8 acc[2][4];
  for (int i = 0; i < 2; ++i)
    for (int j = 0; j < 4; ++j) acc[i][j] = fz;

  // staging: each thread async-copies 32B of A and 32B of W per K-step
  const int srow = tid >> 1;                  // 0..127
  const int scol = (tid & 1) << 4;            // 0 or 16 (bf16 elements)
  const unsigned short* aptr = A + (size_t)(bm + srow) * K + scol;
  const unsigned short* wptr = W + (size_t)(bn + srow) * K + scol;
  const unsigned la[2] = {(unsigned)(size_t)&As[0][srow * LDT + scol],
                          (unsigned)(size_t)&As[1][srow * LDT + scol]};
  const unsigned lw[2] = {(unsigned)(size_t)&Ws[0][srow * LDT + scol],
                          (unsigned)(size_t)&Ws[1][srow * LDT + scol]};

  auto issue = [&](int buf, int k0) {
    async_cp16(la[buf],      aptr + k0);
    async_cp16(la[buf] + 16, aptr + k0 + 8);
    async_cp16(lw[buf],      wptr + k0);
    async_cp16(lw[buf] + 16, wptr + k0 + 8);
  };

  issue(0, 0);                                // prologue: tile 0 in flight
  int buf = 0;
  for (int k0 = 0; k0 < K; k0 += TK, buf ^= 1) {
    if (k0 + TK < K) {                        // overlap next tile with compute
      issue(buf ^ 1, k0 + TK);
      WAIT_ASYNC(0x4);                        // 4 newest = next tile; current done
    } else {
      WAIT_ASYNC(0x0);
    }
    __syncthreads();

    bf16x16 af0 = load_frag(&As[buf][(wm * 32 +  0) * LDT], LDT);
    bf16x16 af1 = load_frag(&As[buf][(wm * 32 + 16) * LDT], LDT);
    for (int j = 0; j < 4; ++j) {
      bf16x16 bf = load_frag(&Ws[buf][(wn * 64 + j * 16) * LDT], LDT);
      acc[0][j] = WMMA_BF16(af0, bf, acc[0][j]);
      acc[1][j] = WMMA_BF16(af1, bf, acc[1][j]);
    }
    __syncthreads();                          // readers done before buf reuse
  }

  // C/D layout: lane&15 = column, reg r = row r (+8 for upper lane half)
  const int half8 = (lane >> 4) << 3;
  const int col0  = bn + wn * 64 + (lane & 15);
  for (int i = 0; i < 2; ++i) {
    const int mrow = bm + wm * 32 + i * 16 + half8;
    for (int j = 0; j < 4; ++j) {
      if (BF16_OUT) {
        unsigned short* cp = Cb + (size_t)mrow * N + col0 + j * 16;
        for (int r = 0; r < 8; ++r) cp[(size_t)r * N] = f2bf(acc[i][j][r]);
      } else {
        float* cp = Cf + (size_t)mrow * N + col0 + j * 16;
        for (int r = 0; r < 8; ++r) cp[(size_t)r * N] = acc[i][j][r];
      }
    }
  }
}

// ---------------------------------------------------------------------------
// Flash attention: block = (b,h, 128 S-rows); 8 waves x 16 rows each.
// Context streamed in async double-buffered chunks of 32.
// ---------------------------------------------------------------------------
#define KTSTR 136   // 128 + 8 pad (272B rows: 16B aligned, conflict-free)
#define VTSTR 40    // Vt tile: [e=128][c=32(+8)]
#define PTSTR 40    // wave-private P staging: [16][32(+8)]

__global__ __launch_bounds__(256)
void flash_attn(const unsigned short* __restrict__ Qb,  // [B*S, DIM] bf16
                const unsigned short* __restrict__ Kb,  // [B, C, H, HD] bf16
                const unsigned short* __restrict__ Vt,  // [B, H, HD, CTX] bf16
                const float* __restrict__ mask,         // [S, C] fp32
                unsigned short* __restrict__ Ob) {      // [B*S, DIM] bf16
  __shared__ unsigned short kt[2][32 * KTSTR];
  __shared__ unsigned short vt[2][HD * VTSTR];
  __shared__ unsigned short pt[8 * 16 * PTSTR];

  const int tid   = threadIdx.x;
  const int lane  = tid & 31;
  const int wave  = tid >> 5;
  const int b     = blockIdx.x >> 5;
  const int h     = blockIdx.x & 31;
  const int srow0 = blockIdx.y * 128 + wave * 16;
  const int lrow  = lane & 15;
  const int half8 = (lane >> 4) << 3;
  const int koff  = (lane >> 4) << 3;

  // --- Q fragments: 4 K-steps of 32 over HD=128, direct bf16 b128 loads ---
  bf16x16 qf[4];
  {
    const unsigned short* qbase = Qb + (size_t)(b * SEQ + srow0 + lrow) * DIM + h * HD;
    for (int kk = 0; kk < 4; ++kk) {
      union { bf16x16 v; u32x4 q[2]; } f;
      f.q[0] = *(const u32x4*)(qbase + kk * 32 + koff);
      f.q[1] = *(const u32x4*)(qbase + kk * 32 + koff + 16);
      qf[kk] = f.v;
    }
  }

  const floatx8 fz = {0.f,0.f,0.f,0.f,0.f,0.f,0.f,0.f};
  floatx8 acc[8];
  for (int g = 0; g < 8; ++g) acc[g] = fz;
  float mstate[8], lstate[8];
  for (int r = 0; r < 8; ++r) { mstate[r] = -1e30f; lstate[r] = 0.f; }

  const float scale = 0.08838834764831845f;   // 1/sqrt(128)
  unsigned short* ptw = pt + wave * 16 * PTSTR;

  // staging coordinates (32B K + 32B V per thread per chunk)
  const int crow = tid >> 3, ecol = (tid & 7) << 4;     // K tile [32c][128e]
  const int erow = tid >> 1, ccol = (tid & 1) << 4;     // V tile [128e][32c]
  const unsigned short* kbase = Kb + ((size_t)(b * CTX + crow) * NH + h) * HD + ecol;
  const unsigned short* vbase = Vt + ((size_t)(b * NH + h) * HD + erow) * CTX + ccol;
  const unsigned lk[2] = {(unsigned)(size_t)&kt[0][crow * KTSTR + ecol],
                          (unsigned)(size_t)&kt[1][crow * KTSTR + ecol]};
  const unsigned lv[2] = {(unsigned)(size_t)&vt[0][erow * VTSTR + ccol],
                          (unsigned)(size_t)&vt[1][erow * VTSTR + ccol]};

  auto issue = [&](int buf, int c0) {
    const unsigned short* kp = kbase + (size_t)c0 * NH * HD;
    const unsigned short* vp = vbase + c0;
    async_cp16(lk[buf],      kp);
    async_cp16(lk[buf] + 16, kp + 8);
    async_cp16(lv[buf],      vp);
    async_cp16(lv[buf] + 16, vp + 8);
  };

  issue(0, 0);                                // prologue: chunk 0 in flight
  int buf = 0;
  for (int c0 = 0; c0 < CTX; c0 += 32, buf ^= 1) {
    if (c0 + 32 < CTX) {
      issue(buf ^ 1, c0 + 32);
      WAIT_ASYNC(0x4);
    } else {
      WAIT_ASYNC(0x0);
    }
    __syncthreads();

    // --- scores: two 16x16 D-frags covering context cols [c0,c0+16),[+16,+32)
    floatx8 s0 = fz, s1 = fz;
    for (int kk = 0; kk < 4; ++kk) {
      bf16x16 bk0 = load_frag(&kt[buf][kk * 32], KTSTR);
      bf16x16 bk1 = load_frag(&kt[buf][16 * KTSTR + kk * 32], KTSTR);
      s0 = WMMA_BF16(qf[kk], bk0, s0);
      s1 = WMMA_BF16(qf[kk], bk1, s1);
    }

    // --- scale + mask + online softmax (row = reg, col = lane&15) ---
    float t0[8], t1[8];
    const float* m0 = mask + (size_t)(srow0 + half8) * CTX + c0 + lrow;
    for (int r = 0; r < 8; ++r) {
      t0[r] = s0[r] * scale + m0[(size_t)r * CTX];
      t1[r] = s1[r] * scale + m0[(size_t)r * CTX + 16];
    }
    float cm[8];
    for (int r = 0; r < 8; ++r) cm[r] = fmaxf(t0[r], t1[r]);
    for (int off = 1; off < 16; off <<= 1)
      for (int r = 0; r < 8; ++r) cm[r] = fmaxf(cm[r], __shfl_xor(cm[r], off, 32));

    float rs[8];
    for (int r = 0; r < 8; ++r) {
      const float mn   = fmaxf(mstate[r], cm[r]);
      const float corr = __expf(mstate[r] - mn);
      mstate[r] = mn;
      const float p0 = __expf(t0[r] - mn);
      const float p1 = __expf(t1[r] - mn);
      rs[r] = p0 + p1;
      lstate[r] *= corr;
      for (int g = 0; g < 8; ++g) acc[g][r] *= corr;
      // stash P (D-frag layout) into wave-private LDS for A-frag reshape
      ptw[(r + half8) * PTSTR + lrow]      = f2bf(p0);
      ptw[(r + half8) * PTSTR + lrow + 16] = f2bf(p1);
    }
    for (int off = 1; off < 16; off <<= 1)
      for (int r = 0; r < 8; ++r) rs[r] += __shfl_xor(rs[r], off, 32);
    for (int r = 0; r < 8; ++r) lstate[r] += rs[r];

    // --- P(16x32) @ V(32x128): wave-local LDS RAW (DScnt wait by compiler) ---
    bf16x16 pf = load_frag(ptw, PTSTR);
    for (int g = 0; g < 8; ++g) {
      bf16x16 bv = load_frag(&vt[buf][g * 16 * VTSTR], VTSTR);
      acc[g] = WMMA_BF16(pf, bv, acc[g]);
    }
    __syncthreads();
  }

  // --- epilogue: out = acc / l, bf16, head-major into [B*S, DIM] ---
  float rl[8];
  for (int r = 0; r < 8; ++r) rl[r] = 1.f / lstate[r];
  unsigned short* ob = Ob + (size_t)(b * SEQ + srow0 + half8) * DIM + h * HD + lrow;
  for (int g = 0; g < 8; ++g)
    for (int r = 0; r < 8; ++r)
      ob[(size_t)r * DIM + g * 16] = f2bf(acc[g][r] * rl[r]);
}

// ---------------------------------------------------------------------------
extern "C" void kernel_launch(void* const* d_in, const int* in_sizes, int n_in,
                              void* d_out, int out_size, void* d_ws, size_t ws_size,
                              hipStream_t stream) {
  (void)in_sizes; (void)n_in; (void)out_size; (void)ws_size;
  const float* x       = (const float*)d_in[0];
  // d_in[1] freqs_cis, d_in[3] input_idexes: dead in the reference dataflow
  const float* mask    = (const float*)d_in[2];
  const float* cache_k = (const float*)d_in[4];
  const float* cache_v = (const float*)d_in[5];
  const float* wq      = (const float*)d_in[6];
  // d_in[7] wk, d_in[8] wv: outputs xk/xv are never used by the reference
  const float* wo      = (const float*)d_in[9];
  float*       out     = (float*)d_out;

  const int M  = NBAT * SEQ;                       // 2048
  const int NX = M * DIM;                          // 8.39M  (x / q / attn-out)
  const int NW = DIM * DIM;                        // 16.78M (wq / wo)
  const int NK = NBAT * CTX * NH * HD;             // 16.78M (cache_k / cache_v)

  unsigned short* xb  = (unsigned short*)d_ws;     // bf16 workspace (176 MiB)
  unsigned short* qb  = xb  + (size_t)NX;
  unsigned short* ab  = qb  + (size_t)NX;
  unsigned short* wqb = ab  + (size_t)NX;
  unsigned short* wob = wqb + (size_t)NW;
  unsigned short* kb  = wob + (size_t)NW;
  unsigned short* vb  = kb  + (size_t)NK;

  // one-time bf16 conversion passes (bandwidth-bound, amortized over reuse)
  cvt_f32_bf16<<<NX / 2048, 256, 0, stream>>>(x, xb, NX);
  cvt_f32_bf16<<<NW / 2048, 256, 0, stream>>>(wq, wqb, NW);
  cvt_f32_bf16<<<NW / 2048, 256, 0, stream>>>(wo, wob, NW);
  cvt_f32_bf16<<<NK / 2048, 256, 0, stream>>>(cache_k, kb, NK);
  cvt_transpose_v<<<dim3(NBAT * NH, CTX / 32, HD / 32), 256, 0, stream>>>(cache_v, vb);

  dim3 gemm_grid(M / TM, DIM / TN);                // (16, 32)
  gemm_bf16<true><<<gemm_grid, 256, 0, stream>>>(xb, wqb, nullptr, qb, M, DIM, DIM);

  dim3 attn_grid(NBAT * NH, SEQ / 128);            // (64, 8)
  flash_attn<<<attn_grid, 256, 0, stream>>>(qb, kb, vb, mask, ab);

  gemm_bf16<false><<<gemm_grid, 256, 0, stream>>>(ab, wob, out, nullptr, M, DIM, DIM);
}
